// Attention_62809601736889
// MI455X (gfx1250) — compile-verified, matrix operand order checked by reading, not run on hardware
//
#include <hip/hip_runtime.h>
#include <hip/hip_bf16.h>

typedef __bf16 bf16_t;
typedef __attribute__((ext_vector_type(16))) __bf16 v16bf;
typedef __attribute__((ext_vector_type(8)))  __bf16 v8bf;
typedef __attribute__((ext_vector_type(4)))  __bf16 v4bf;
typedef __attribute__((ext_vector_type(8)))  float  v8f;

#define DIM   768
#define HEADS 12
#define HD    64
#define SEQ   2048
#define BATCH 2
#define ROWS  (BATCH * SEQ)   // 4096
#define EQKV  (3 * DIM)       // 2304

// Load a v16bf A/B fragment per the CDNA5 16-bit 16x32 layout:
// lane (m + 16*kh) holds K = koff..koff+7 (elems 0-7) and koff+16..koff+23
// (elems 8-15), koff = 8*kh.  Caller passes pointer at (row, k0+koff).
static __device__ __forceinline__ v16bf load_ab_frag(const bf16_t* p) {
  v8bf lo = *(const v8bf*)(p);
  v8bf hi = *(const v8bf*)(p + 16);
  v16bf f;
#pragma unroll
  for (int i = 0; i < 8; ++i) { f[i] = lo[i]; f[i + 8] = hi[i]; }
  return f;
}

static __device__ __forceinline__ v8f wmma_bf16(v16bf a, v16bf b, v8f c) {
  // (neg_a, A, neg_b, B, c_mod, C, reuse_a, reuse_b)
  return __builtin_amdgcn_wmma_f32_16x16x32_bf16(false, a, false, b,
                                                 (short)0, c, false, false);
}

// ---------------------------------------------------------------- convert
__global__ __launch_bounds__(256) void cvt_f32_bf16(const float* __restrict__ in,
                                                    bf16_t* __restrict__ out,
                                                    int n4) {
  int i = blockIdx.x * 256 + threadIdx.x;
  if (i < n4) {
    float4 v = ((const float4*)in)[i];
    v4bf o;
    o[0] = (bf16_t)v.x; o[1] = (bf16_t)v.y;
    o[2] = (bf16_t)v.z; o[3] = (bf16_t)v.w;
    ((v4bf*)out)[i] = o;
  }
}

// ---------------------------------------------------------------- QKV GEMM
// qkv[bn, e] = sum_d x[bn,d] * w_qkv[e,d].  A = x_bf16 [4096,768] row-major,
// B columns = w_qkv rows (K contiguous) -> both frag loads are b128s.
// Epilogue scatters Q,K as [B,H,N,64] bf16 and V transposed as [B,H,64,N].
__global__ __launch_bounds__(256) void qkv_gemm(const bf16_t* __restrict__ A,
                                                const bf16_t* __restrict__ B,
                                                bf16_t* __restrict__ Q,
                                                bf16_t* __restrict__ Kd,
                                                bf16_t* __restrict__ Vt) {
  const int lane = threadIdx.x & 31;
  const int wave = threadIdx.x >> 5;
  const int NB   = EQKV / 256;              // 9 column blocks
  const int bm   = blockIdx.x / NB;
  const int bn   = blockIdx.x % NB;
  const int m0   = bm * 128 + (wave >> 2) * 64;
  const int n0   = bn * 256 + (wave & 3) * 64;
  const int r    = lane & 15;
  const int hlf  = lane >> 4;
  const int koff = hlf << 3;

  v8f acc[4][4] = {};
  const bf16_t* Ap = A + (size_t)(m0 + r) * DIM + koff;
  const bf16_t* Bp = B + (size_t)(n0 + r) * DIM + koff;

  for (int k0 = 0; k0 < DIM; k0 += 32) {
    __builtin_prefetch(Ap + k0 + 128, 0, 0);
    __builtin_prefetch(Bp + k0 + 128, 0, 0);
    v16bf af[4], bf[4];
#pragma unroll
    for (int t = 0; t < 4; ++t) af[t] = load_ab_frag(Ap + (size_t)t * 16 * DIM + k0);
#pragma unroll
    for (int t = 0; t < 4; ++t) bf[t] = load_ab_frag(Bp + (size_t)t * 16 * DIM + k0);
#pragma unroll
    for (int i = 0; i < 4; ++i)
#pragma unroll
      for (int j = 0; j < 4; ++j)
        acc[i][j] = wmma_bf16(af[i], bf[j], acc[i][j]);
  }

  // C layout: elem i of v8f at lane L is row (i + 8*(L>>4)), col (L&15).
#pragma unroll
  for (int tn = 0; tn < 4; ++tn) {
    const int e_base = n0 + tn * 16;          // 64-aligned tiles => head-uniform
    const int qkv    = e_base / DIM;
    const int h      = (e_base % DIM) / HD;
    const int d      = (e_base % HD) + r;
    bf16_t* dst = (qkv == 0) ? Q : (qkv == 1) ? Kd : Vt;
#pragma unroll
    for (int tm = 0; tm < 4; ++tm) {
#pragma unroll
      for (int i = 0; i < 8; ++i) {
        const int rg = m0 + tm * 16 + i + 8 * hlf;
        const int bb = rg >> 11;
        const int ns = rg & (SEQ - 1);
        const size_t off = (qkv < 2)
            ? ((size_t)(bb * HEADS + h) * SEQ + ns) * HD + d
            : ((size_t)(bb * HEADS + h) * HD + d) * SEQ + ns;
        dst[off] = (bf16_t)acc[tm][tn][i];
      }
    }
  }
}

// ---------------------------------------------------------------- flash attn
// One wave = 16 query rows.  Scores never touch HBM (400 MB avoided).
__global__ __launch_bounds__(256) void flash_attn(const bf16_t* __restrict__ Q,
                                                  const bf16_t* __restrict__ K,
                                                  const bf16_t* __restrict__ Vt,
                                                  bf16_t* __restrict__ O) {
  __shared__ bf16_t pstage[8][16][40];   // per-wave slab, row pad -> no bank clash
  const int lane = threadIdx.x & 31;
  const int wave = threadIdx.x >> 5;
  const int bh   = blockIdx.x >> 4;      // 0..23 = b*12+h
  const int qt   = blockIdx.x & 15;
  const int b    = bh / HEADS;
  const int h    = bh % HEADS;
  const int r    = lane & 15;
  const int hlf  = lane >> 4;
  const int koff = hlf << 3;
  const int qbase = qt * 128 + wave * 16;

  const bf16_t* Qb = Q  + (size_t)bh * SEQ * HD;
  const bf16_t* Kb = K  + (size_t)bh * SEQ * HD;
  const bf16_t* Vb = Vt + (size_t)bh * HD * SEQ;

  v16bf qa[2];
#pragma unroll
  for (int t = 0; t < 2; ++t)
    qa[t] = load_ab_frag(Qb + (size_t)(qbase + r) * HD + t * 32 + koff);

  float mrun[8], lrun[8];
#pragma unroll
  for (int i = 0; i < 8; ++i) { mrun[i] = -1e30f; lrun[i] = 0.0f; }
  v8f o[4] = {};
  const float scale = 0.125f;            // 64^-0.5

  for (int j0 = 0; j0 < SEQ; j0 += 32) {
    v8f s0 = {}, s1 = {};                // S tile 16x32, two 16x16 subtiles
#pragma unroll
    for (int t = 0; t < 2; ++t) {        // K-dim (= head dim) in two 32-steps
      v16bf kb0 = load_ab_frag(Kb + (size_t)(j0 + r)      * HD + t * 32 + koff);
      v16bf kb1 = load_ab_frag(Kb + (size_t)(j0 + 16 + r) * HD + t * 32 + koff);
      s0 = wmma_bf16(qa[t], kb0, s0);
      s1 = wmma_bf16(qa[t], kb1, s1);
    }

    float alpha[8];
#pragma unroll
    for (int i = 0; i < 8; ++i) {        // online softmax, per accumulator row
      float a0 = s0[i] * scale, a1 = s1[i] * scale;
      float rm = fmaxf(a0, a1);
#pragma unroll
      for (int m = 1; m <= 8; m <<= 1)   // xor<=8 stays inside 16-lane half
        rm = fmaxf(rm, __shfl_xor(rm, m, 32));
      const float mn = fmaxf(mrun[i], rm);
      const float p0 = __expf(a0 - mn), p1 = __expf(a1 - mn);
      float rs = p0 + p1;
#pragma unroll
      for (int m = 1; m <= 8; m <<= 1) rs += __shfl_xor(rs, m, 32);
      alpha[i] = __expf(mrun[i] - mn);
      lrun[i]  = lrun[i] * alpha[i] + rs;
      mrun[i]  = mn;
      s0[i] = p0; s1[i] = p1;
    }
#pragma unroll
    for (int t = 0; t < 4; ++t)
#pragma unroll
      for (int i = 0; i < 8; ++i) o[t][i] *= alpha[i];

    // Re-shape P from C-layout to A-fragment layout via per-wave LDS slab.
#pragma unroll
    for (int i = 0; i < 8; ++i) {
      const int row = i + 8 * hlf;
      pstage[wave][row][r]      = (bf16_t)s0[i];
      pstage[wave][row][16 + r] = (bf16_t)s1[i];
    }
    __builtin_amdgcn_wave_barrier();     // lockstep wave, LDS in-order: no s_barrier
    v16bf pa = load_ab_frag(&pstage[wave][r][koff]);
    __builtin_amdgcn_wave_barrier();

    // O += P * V ; V stored transposed so B-frag columns are contiguous keys.
#pragma unroll
    for (int t = 0; t < 4; ++t) {
      v16bf vb = load_ab_frag(Vb + (size_t)(t * 16 + r) * SEQ + j0 + koff);
      o[t] = wmma_bf16(pa, vb, o[t]);
    }
  }

#pragma unroll
  for (int t = 0; t < 4; ++t)
#pragma unroll
    for (int i = 0; i < 8; ++i) {
      const int ns = qbase + i + 8 * hlf;
      const float v = o[t][i] / lrun[i];
      O[((size_t)(b * SEQ + ns)) * DIM + h * HD + t * 16 + r] = (bf16_t)v;
    }
}

// ---------------------------------------------------------------- output proj
__global__ __launch_bounds__(256) void proj_gemm(const bf16_t* __restrict__ A,
                                                 const bf16_t* __restrict__ B,
                                                 const float* __restrict__ bias,
                                                 float* __restrict__ out) {
  const int lane = threadIdx.x & 31;
  const int wave = threadIdx.x >> 5;
  const int NB   = DIM / 256;               // 3 column blocks
  const int bm   = blockIdx.x / NB;
  const int bn   = blockIdx.x % NB;
  const int m0   = bm * 128 + (wave >> 2) * 64;
  const int n0   = bn * 256 + (wave & 3) * 64;
  const int r    = lane & 15;
  const int hlf  = lane >> 4;
  const int koff = hlf << 3;

  v8f acc[4][4] = {};
  const bf16_t* Ap = A + (size_t)(m0 + r) * DIM + koff;
  const bf16_t* Bp = B + (size_t)(n0 + r) * DIM + koff;

  for (int k0 = 0; k0 < DIM; k0 += 32) {
    __builtin_prefetch(Ap + k0 + 128, 0, 0);
    __builtin_prefetch(Bp + k0 + 128, 0, 0);
    v16bf af[4], bf[4];
#pragma unroll
    for (int t = 0; t < 4; ++t) af[t] = load_ab_frag(Ap + (size_t)t * 16 * DIM + k0);
#pragma unroll
    for (int t = 0; t < 4; ++t) bf[t] = load_ab_frag(Bp + (size_t)t * 16 * DIM + k0);
#pragma unroll
    for (int i = 0; i < 4; ++i)
#pragma unroll
      for (int j = 0; j < 4; ++j)
        acc[i][j] = wmma_bf16(af[i], bf[j], acc[i][j]);
  }

#pragma unroll
  for (int tn = 0; tn < 4; ++tn) {
    const int cg = n0 + tn * 16 + r;
    const float bv = bias[cg];
#pragma unroll
    for (int tm = 0; tm < 4; ++tm)
#pragma unroll
      for (int i = 0; i < 8; ++i) {
        const int rg = m0 + tm * 16 + i + 8 * hlf;
        out[(size_t)rg * DIM + cg] = acc[tm][tn][i] + bv;
      }
  }
}

// ---------------------------------------------------------------- launch
extern "C" void kernel_launch(void* const* d_in, const int* in_sizes, int n_in,
                              void* d_out, int out_size, void* d_ws, size_t ws_size,
                              hipStream_t stream) {
  const float* x      = (const float*)d_in[0];
  const float* w_qkv  = (const float*)d_in[1];
  const float* w_proj = (const float*)d_in[2];
  const float* b_proj = (const float*)d_in[3];
  float* out = (float*)d_out;

  // workspace layout (bf16 elements, all 16B-aligned): ~34.5 MB total
  bf16_t* xb   = (bf16_t*)d_ws;
  bf16_t* wqb  = xb  + (size_t)ROWS * DIM;
  bf16_t* wpb  = wqb + (size_t)EQKV * DIM;
  bf16_t* Q    = wpb + (size_t)DIM * DIM;
  bf16_t* Kd   = Q   + (size_t)ROWS * DIM;   // H*HD == DIM
  bf16_t* Vt   = Kd  + (size_t)ROWS * DIM;
  bf16_t* Obuf = Vt  + (size_t)ROWS * DIM;

  int n;
  n = ROWS * DIM / 4; cvt_f32_bf16<<<(n + 255) / 256, 256, 0, stream>>>(x, xb, n);
  n = EQKV * DIM / 4; cvt_f32_bf16<<<(n + 255) / 256, 256, 0, stream>>>(w_qkv, wqb, n);
  n = DIM * DIM / 4;  cvt_f32_bf16<<<(n + 255) / 256, 256, 0, stream>>>(w_proj, wpb, n);

  qkv_gemm<<<(ROWS / 128) * (EQKV / 256), 256, 0, stream>>>(xb, wqb, Q, Kd, Vt);
  flash_attn<<<BATCH * HEADS * (SEQ / 128), 256, 0, stream>>>(Q, Kd, Vt, Obuf);
  proj_gemm<<<(ROWS / 128) * (DIM / 256), 256, 0, stream>>>(Obuf, wpb, b_proj, out);
}
